// GNN_8950711845985
// MI455X (gfx1250) — matrix operand order repaired and checked by reading, not compiled, not run on hardware
//
#include <hip/hip_runtime.h>

// ---------------------------------------------------------------------------
// Types for CDNA5 WMMA (wave32, v_wmma_f32_16x16x32_bf16)
// ---------------------------------------------------------------------------
typedef __bf16 bf16_t;
typedef __attribute__((ext_vector_type(16))) __bf16 v16bf;
typedef __attribute__((ext_vector_type(8)))  float  v8f;

union Frag16 {            // 16 bf16 = 32 bytes = two 16B chunks
    v16bf v;
    uint4 q[2];
};

static __device__ __forceinline__ v8f wmma_bf16(v16bf a, v16bf b, v8f c) {
    // (neg_a, A, neg_b, B, c_mod, C, reuse_a, reuse_b)
    return __builtin_amdgcn_wmma_f32_16x16x32_bf16(false, a, false, b,
                                                   (short)0, c, false, false);
}

static __device__ __forceinline__ unsigned short f2bf(float f) {
    unsigned u = __float_as_uint(f);
    unsigned r = u + 0x7FFFu + ((u >> 16) & 1u);   // round-to-nearest-even
    return (unsigned short)(r >> 16);
}
static __device__ __forceinline__ float bf2f(unsigned short h) {
    return __uint_as_float(((unsigned)h) << 16);
}
static __device__ __forceinline__ v8f zero8() {
    v8f z;
#pragma unroll
    for (int i = 0; i < 8; ++i) z[i] = 0.0f;
    return z;
}

#define BATCH 512
#define NNODE 128
#define DIM   512
#define LPAD  520   // LDS row stride in halves: 1040B, 16B-aligned, banks spread 4
#define MROWS 32    // node rows per k_sage block

// ---------------------------------------------------------------------------
// K0: f32 -> bf16 + reciprocal L2 norms (one block per node row)
// ---------------------------------------------------------------------------
__global__ void k_norm(const float* __restrict__ x,
                       unsigned short* __restrict__ xbf,
                       float* __restrict__ rnorm) {
    __shared__ float red[256];
    const int row  = blockIdx.x;            // 0 .. B*N-1
    const int t    = threadIdx.x;
    const int base = row * DIM;
    float a0 = x[base + t];
    float a1 = x[base + t + 256];
    xbf[base + t]       = f2bf(a0);
    xbf[base + t + 256] = f2bf(a1);
    red[t] = a0 * a0 + a1 * a1;
    __syncthreads();
    for (int s = 128; s > 0; s >>= 1) {
        if (t < s) red[t] += red[t + s];
        __syncthreads();
    }
    if (t == 0) rnorm[row] = rsqrtf(red[0] + 1e-12f);
}

// ---------------------------------------------------------------------------
// K1: cosine kNN (top-3 incl. self) per batch, Gram matrix via bf16 WMMA
// ---------------------------------------------------------------------------
__global__ void k_knn(const unsigned short* __restrict__ xbf,
                      const float* __restrict__ rnorm,
                      int* __restrict__ idx) {
    __shared__ float sim[NNODE][132];       // 128x132 f32, padded columns
    const int b    = blockIdx.x;
    const int tid  = threadIdx.x;
    const int w    = tid >> 5;              // wave id 0..7 -> node-row tile
    const int lane = tid & 31;
    const int half = lane >> 4;
    const int l16  = lane & 15;
    const unsigned short* xb = xbf + (size_t)b * NNODE * DIM;

    const int nt = w;                                   // rows n = nt*16..+15
    const unsigned short* arow = xb + (nt * 16 + l16) * DIM;

    for (int mt = 0; mt < 8; ++mt) {
        v8f acc = zero8();
        const unsigned short* brow = xb + (mt * 16 + l16) * DIM;
#pragma unroll 4
        for (int kt = 0; kt < 16; ++kt) {
            Frag16 a, bf;
            const int d0 = kt * 32;
            a.q[0]  = *(const uint4*)(arow + d0 + half * 16);
            a.q[1]  = *(const uint4*)(arow + d0 + half * 16 + 8);
            const unsigned short* bp = brow + d0 + half * 16;
            bf.q[0] = ((const uint4*)bp)[0];
            bf.q[1] = ((const uint4*)bp)[1];
            acc = wmma_bf16(a.v, bf.v, acc);
        }
        const int   m  = mt * 16 + l16;                 // column owned by lane
        const float rm = rnorm[b * NNODE + m];
#pragma unroll
        for (int r = 0; r < 8; ++r) {
            const int n = nt * 16 + r + 8 * half;       // C/D row layout
            sim[n][m] = acc[r] * rm * rnorm[b * NNODE + n];
        }
    }
    __syncthreads();

    if (tid < NNODE) {
        const int n = tid;
        float v0 = -3.4e38f, v1 = -3.4e38f, v2 = -3.4e38f;
        int   i0 = 0,        i1 = 0,        i2 = 0;
        for (int m = 0; m < NNODE; ++m) {
            float s = sim[n][m];
            if (s > v0)      { v2=v1; i2=i1; v1=v0; i1=i0; v0=s; i0=m; }
            else if (s > v1) { v2=v1; i2=i1; v1=s; i1=m; }
            else if (s > v2) { v2=s; i2=m; }
        }
        int* ip = idx + ((size_t)b * NNODE + n) * 3;
        ip[0] = i0; ip[1] = i1; ip[2] = i2;
    }
}

// ---------------------------------------------------------------------------
// K2: pack a 512x512 f32 weight (O x D, out = act @ W^T) into the WMMA
//     B-operand lane layout:  P[((kt*32+ot)*32+lane)*16 + e]
//       = W[ot*16 + lane%16][kt*32 + (lane/16)*16 + e]
// ---------------------------------------------------------------------------
__global__ void k_pack(const float* __restrict__ W,
                       unsigned short* __restrict__ P) {
    const int p    = blockIdx.x * 256 + threadIdx.x;    // 0 .. 512*512-1
    const int e    = p & 15;
    const int lane = (p >> 4) & 31;
    const int ot   = (p >> 9) & 31;
    const int kt   = p >> 14;
    const int o    = ot * 16 + (lane & 15);
    const int d    = kt * 32 + (lane >> 4) * 16 + e;
    P[p] = f2bf(W[o * DIM + d]);
}

// ---------------------------------------------------------------------------
// K3: SAGEConv layer: out = aggr @ Wl^T + x @ Wr^T + bl  (+ optional leaky)
//     Block = (batch, 32-node tile); 8 waves x (2 n-tiles x 4 o-tiles) each.
//     Each B fragment feeds two back-to-back WMMAs (shared across n-tiles)
//     -> 2x arithmetic intensity vs. L2, half the weight traffic.
// ---------------------------------------------------------------------------
__global__ void k_sage(const unsigned short* __restrict__ xin,
                       const int* __restrict__ idx,
                       const unsigned short* __restrict__ Pl,
                       const unsigned short* __restrict__ Pr,
                       const float* __restrict__ bl,
                       unsigned short* __restrict__ xout,
                       int leaky) {
    __shared__ unsigned short sx[MROWS][LPAD];   // self rows      (bf16)
    __shared__ unsigned short sa[MROWS][LPAD];   // neighbor means (bf16)
    __shared__ int nb[MROWS][3];

    const int b   = blockIdx.x >> 2;             // NNODE/MROWS = 4 tiles
    const int nt  = blockIdx.x & 3;
    const int nb0 = nt * MROWS;
    const int tid = threadIdx.x;

    if (tid < MROWS * 3) {
        nb[tid / 3][tid % 3] = idx[((size_t)b * NNODE + nb0 + tid / 3) * 3 + tid % 3];
    }
    __syncthreads();

    const unsigned short* xbat = xin + (size_t)b * NNODE * DIM;
    for (int i = tid; i < MROWS * DIM; i += 256) {
        const int r = i >> 9;
        const int d = i & (DIM - 1);
        sx[r][d] = xbat[(nb0 + r) * DIM + d];
        float s = bf2f(xbat[nb[r][0] * DIM + d])
                + bf2f(xbat[nb[r][1] * DIM + d])
                + bf2f(xbat[nb[r][2] * DIM + d]);
        sa[r][d] = f2bf(s * (1.0f / 3.0f));
    }
    __syncthreads();

    const int w    = tid >> 5;     // wave 0..7 -> o-columns [w*64, w*64+64)
    const int lane = tid & 31;
    const int half = lane >> 4;
    const int l16  = lane & 15;

    v8f acc[2][4];
#pragma unroll
    for (int h = 0; h < 2; ++h)
#pragma unroll
        for (int j = 0; j < 4; ++j) acc[h][j] = zero8();

    for (int kt = 0; kt < 16; ++kt) {
        // prefetch next k-step's weight fragments into cache (global_prefetch_b8)
        if (kt < 15) {
            const size_t pn = ((size_t)((kt + 1) * 32 + w * 4) * 32 + lane) * 16;
            __builtin_prefetch(Pl + pn, 0, 0);
            __builtin_prefetch(Pr + pn, 0, 0);
        }
        Frag16 ax[2], aa[2];
        const int d0 = kt * 32 + half * 16;
#pragma unroll
        for (int h = 0; h < 2; ++h) {
            const int row = h * 16 + l16;              // A rows: n = row
            ax[h].q[0] = *(const uint4*)(&sx[row][d0]);
            ax[h].q[1] = *(const uint4*)(&sx[row][d0 + 8]);
            aa[h].q[0] = *(const uint4*)(&sa[row][d0]);
            aa[h].q[1] = *(const uint4*)(&sa[row][d0 + 8]);
        }
#pragma unroll
        for (int j = 0; j < 4; ++j) {
            const int ot = w * 4 + j;
            const size_t poff = ((size_t)(kt * 32 + ot) * 32 + lane) * 16;
            Frag16 fl, fr;
            const uint4* pl = (const uint4*)(Pl + poff);
            fl.q[0] = pl[0]; fl.q[1] = pl[1];
            const uint4* pr = (const uint4*)(Pr + poff);
            fr.q[0] = pr[0]; fr.q[1] = pr[1];
            // share each B operand across both n-tiles (consecutive WMMAs)
            acc[0][j] = wmma_bf16(aa[0].v, fl.v, acc[0][j]);
            acc[1][j] = wmma_bf16(aa[1].v, fl.v, acc[1][j]);
            acc[0][j] = wmma_bf16(ax[0].v, fr.v, acc[0][j]);
            acc[1][j] = wmma_bf16(ax[1].v, fr.v, acc[1][j]);
        }
    }

    unsigned short* obat = xout + (size_t)b * NNODE * DIM;
#pragma unroll
    for (int h = 0; h < 2; ++h) {
#pragma unroll
        for (int j = 0; j < 4; ++j) {
            const int   o    = (w * 4 + j) * 16 + l16;
            const float bias = bl[o];
#pragma unroll
            for (int r = 0; r < 8; ++r) {
                float v = acc[h][j][r] + bias;
                if (leaky) v = (v >= 0.0f) ? v : 0.2f * v;
                const int n = nb0 + h * 16 + r + 8 * half;   // C/D row layout
                obat[n * DIM + o] = f2bf(v);
            }
        }
    }
}

// ---------------------------------------------------------------------------
// K4: global mean pool over nodes + 2-way linear classifier (f32 out)
// ---------------------------------------------------------------------------
__global__ void k_pool(const unsigned short* __restrict__ x3,
                       const float* __restrict__ wc,
                       const float* __restrict__ bc,
                       float* __restrict__ out) {
    __shared__ float r0[256];
    __shared__ float r1[256];
    const int b   = blockIdx.x;
    const int tid = threadIdx.x;
    const unsigned short* xb = x3 + (size_t)b * NNODE * DIM;
    float p0 = 0.0f, p1 = 0.0f;
    for (int d = tid; d < DIM; d += 256) {
        float s = 0.0f;
        for (int n = 0; n < NNODE; ++n) s += bf2f(xb[n * DIM + d]);
        const float h = s * (1.0f / 128.0f);
        p0 += h * wc[d];
        p1 += h * wc[DIM + d];
    }
    r0[tid] = p0; r1[tid] = p1;
    __syncthreads();
    for (int s = 128; s > 0; s >>= 1) {
        if (tid < s) { r0[tid] += r0[tid + s]; r1[tid] += r1[tid + s]; }
        __syncthreads();
    }
    if (tid == 0) {
        out[b * 2 + 0] = r0[0] + bc[0];
        out[b * 2 + 1] = r1[0] + bc[1];
    }
}

// ---------------------------------------------------------------------------
// Host launcher
// ---------------------------------------------------------------------------
extern "C" void kernel_launch(void* const* d_in, const int* in_sizes, int n_in,
                              void* d_out, int out_size, void* d_ws, size_t ws_size,
                              hipStream_t stream) {
    (void)in_sizes; (void)n_in; (void)out_size; (void)ws_size;

    const float* enc = (const float*)d_in[0];
    const float* w1l = (const float*)d_in[1];
    const float* b1l = (const float*)d_in[2];
    const float* w1r = (const float*)d_in[3];
    const float* w2l = (const float*)d_in[4];
    const float* b2l = (const float*)d_in[5];
    const float* w2r = (const float*)d_in[6];
    const float* w3l = (const float*)d_in[7];
    const float* b3l = (const float*)d_in[8];
    const float* w3r = (const float*)d_in[9];
    const float* wc  = (const float*)d_in[10];
    const float* bc  = (const float*)d_in[11];
    float* out = (float*)d_out;

    // workspace layout (bytes)
    unsigned char* ws = (unsigned char*)d_ws;
    const size_t SZ_ACT = (size_t)BATCH * NNODE * DIM * 2;      // 64 MiB bf16
    unsigned short* xbf  = (unsigned short*)(ws);                // acts buf A
    unsigned short* abuf = (unsigned short*)(ws + SZ_ACT);       // acts buf B
    int*   idx   = (int*)  (ws + 2 * SZ_ACT);                    // B*N*3 int
    float* rnorm = (float*)(ws + 2 * SZ_ACT + (size_t)BATCH * NNODE * 3 * 4);
    unsigned short* pw = (unsigned short*)(ws + 2 * SZ_ACT
                         + (size_t)BATCH * NNODE * 3 * 4
                         + (size_t)BATCH * NNODE * 4);
    const size_t WSTEP = (size_t)DIM * DIM;                      // 256K halves
    unsigned short* P1l = pw + 0 * WSTEP;
    unsigned short* P1r = pw + 1 * WSTEP;
    unsigned short* P2l = pw + 2 * WSTEP;
    unsigned short* P2r = pw + 3 * WSTEP;
    unsigned short* P3l = pw + 4 * WSTEP;
    unsigned short* P3r = pw + 5 * WSTEP;

    // 0) bf16 convert + norms
    k_norm<<<BATCH * NNODE, 256, 0, stream>>>(enc, xbf, rnorm);

    // 1) per-batch cosine kNN via WMMA Gram matrix
    k_knn<<<BATCH, 256, 0, stream>>>(xbf, rnorm, idx);

    // 2) pack all six GEMM weights into WMMA B-fragment layout
    const int PGRID = (DIM * DIM) / 256;
    k_pack<<<PGRID, 256, 0, stream>>>(w1l, P1l);
    k_pack<<<PGRID, 256, 0, stream>>>(w1r, P1r);
    k_pack<<<PGRID, 256, 0, stream>>>(w2l, P2l);
    k_pack<<<PGRID, 256, 0, stream>>>(w2r, P2r);
    k_pack<<<PGRID, 256, 0, stream>>>(w3l, P3l);
    k_pack<<<PGRID, 256, 0, stream>>>(w3r, P3r);

    // 3) three SAGE layers, ping-pong activations
    const int SGRID = BATCH * (NNODE / MROWS);
    k_sage<<<SGRID, 256, 0, stream>>>(xbf,  idx, P1l, P1r, b1l, abuf, 1);
    k_sage<<<SGRID, 256, 0, stream>>>(abuf, idx, P2l, P2r, b2l, xbf,  1);
    k_sage<<<SGRID, 256, 0, stream>>>(xbf,  idx, P3l, P3r, b3l, abuf, 0);

    // 4) mean pool + classifier
    k_pool<<<BATCH, 256, 0, stream>>>(abuf, wc, bc, out);
}
